// MultiBoxLoss_10350871183831
// MI455X (gfx1250) — matrix operand order repaired and checked by reading, not compiled, not run on hardware
//
#include <hip/hip_runtime.h>

typedef __attribute__((ext_vector_type(16))) __bf16 v16bf;
typedef __attribute__((ext_vector_type(8)))  float  v8f;

#define Bn    8
#define Pn    65536
#define Cn    21
#define Hn    128
#define Wdn   128
#define An    4
#define EMBn  256
#define NIDn  1000
#define NPAD  1008          // 63 tiles of 16 classes
#define NT    (NPAD/16)
#define LOCn  (Bn*Hn*Wdn)   // 131072 rows
#define EMB_SCALE 9.7676262f // sqrt(2)*log(999)

#define ROWS_PER_BLOCK 128
#define WAVES 8
#define AK 264              // padded LDS stride (bf16 elems) to spread banks

__device__ __forceinline__ unsigned short f2bf(float f) {
    unsigned int u = __float_as_uint(f);
    unsigned int r = (u + 0x7FFFu + ((u >> 16) & 1u)) >> 16; // RNE
    return (unsigned short)r;
}

__device__ __forceinline__ void wait_async0() {
#if __has_builtin(__builtin_amdgcn_s_wait_asynccnt)
    __builtin_amdgcn_s_wait_asynccnt(0);
#else
    asm volatile("s_wait_asynccnt 0x0" ::: "memory");
#endif
}

// ---------------- init: zero integer counters -------------------------------
__global__ void k_init(int* cnt) {
    int t = threadIdx.x;
    if (t < 17) cnt[t] = 0;   // numPos[8], tieCnt[8], validCount
}

// ---------------- bg_loss -> monotone keys, count positives -----------------
__global__ __launch_bounds__(256)
void k_bgloss(const float* __restrict__ conf, const int* __restrict__ labels,
              unsigned int* __restrict__ keys, int* __restrict__ numPos) {
    int idx = blockIdx.x * 256 + threadIdx.x;     // [0, B*P)
    if (idx >= Bn * Pn) return;
    const float* c = conf + (size_t)idx * Cn;
    float m = c[0];
    #pragma unroll
    for (int j = 1; j < Cn; ++j) m = fmaxf(m, c[j]);
    float s = 0.f;
    #pragma unroll
    for (int j = 0; j < Cn; ++j) s += __expf(c[j] - m);
    float bg = m + __logf(s) - c[0];              // -log_softmax[...,0]
    int lab = labels[idx];
    bool pos = lab > 0;
    unsigned int key;
    if (pos) key = 0u;                            // excluded from negative pool
    else {
        unsigned int bits = __float_as_uint(bg);
        key = (bits & 0x80000000u) ? ~bits : (bits | 0x80000000u); // order-preserving
    }
    keys[idx] = key;
    unsigned long long bal = __ballot(pos);
    if ((threadIdx.x & 31) == 0) {
        int cpos = __popcll(bal);
        if (cpos) atomicAdd(&numPos[idx >> 16], cpos);  // P = 65536
    }
}

// ---------------- W (f32) -> bf16, padded to 1008 rows ----------------------
__global__ __launch_bounds__(256)
void k_wconv(const float* __restrict__ Wm, unsigned short* __restrict__ wbf) {
    int idx = blockIdx.x * 256 + threadIdx.x;     // [0, 1008*256)
    int n = idx >> 8;
    float v = (n < NIDn) ? Wm[idx] : 0.f;
    wbf[idx] = f2bf(v);
}

// ---------------- exact radix top-k select per batch ------------------------
__global__ __launch_bounds__(1024)
void k_radix(const unsigned int* __restrict__ keys, const int* __restrict__ numPos,
             unsigned int* __restrict__ Tarr, int* __restrict__ Rarr) {
    __shared__ int hist[256];
    __shared__ unsigned int s_prefix;
    __shared__ int s_k;
    int b = blockIdx.x, tid = threadIdx.x;
    if (tid == 0) { s_prefix = 0u; s_k = numPos[b] * 15; }
    __syncthreads();
    if (s_k <= 0) {
        if (tid == 0) { Tarr[b] = 0xFFFFFFFFu; Rarr[b] = 0; }
        return;
    }
    for (int pass = 0; pass < 4; ++pass) {
        int shift = 24 - 8 * pass;
        if (tid < 256) hist[tid] = 0;
        __syncthreads();
        unsigned int pref = s_prefix;
        for (int i = tid; i < Pn; i += 1024) {
            unsigned int k = keys[(size_t)b * Pn + i];
            bool match = (pass == 0) ? true
                : ((k >> (shift + 8)) == (pref >> (shift + 8)));
            if (match) atomicAdd(&hist[(k >> shift) & 255], 1);
        }
        __syncthreads();
        if (tid == 0) {
            int kk = s_k, acc = 0, bin;
            for (bin = 255; bin >= 0; --bin) {
                if (acc + hist[bin] >= kk) break;
                acc += hist[bin];
            }
            if (bin < 0) bin = 0;
            s_prefix = pref | (((unsigned int)bin) << shift);
            s_k = kk - acc;                        // needed among == this byte
        }
        __syncthreads();
    }
    if (tid == 0) { Tarr[b] = s_prefix; Rarr[b] = s_k; }
}

// ---------------- mask|gid per location (anchor max), count valid -----------
__global__ __launch_bounds__(256)
void k_mask(const unsigned int* __restrict__ keys, const int* __restrict__ labels,
            const int* __restrict__ gt, const unsigned int* __restrict__ Tarr,
            const int* __restrict__ Rarr, int* __restrict__ tieCnt,
            int* __restrict__ gidmask, int* __restrict__ validCount) {
    int idx = blockIdx.x * 256 + threadIdx.x;     // [0, LOCn)
    int b = idx >> 14;                            // H*W = 16384
    unsigned int T = Tarr[b];
    int R = Rarr[b];
    bool any = false;
    int g = 0;                                    // gt_ids >= 0 always
    #pragma unroll
    for (int a = 0; a < An; ++a) {
        int p = idx * An + a;
        unsigned int k = keys[p];
        bool pos = labels[p] > 0;
        bool sel = pos;
        if (!pos) {
            if (k > T) sel = true;
            else if (k == T && R > 0) sel = (atomicAdd(&tieCnt[b], 1) < R);
        }
        any = any || sel;
        g = max(g, gt[p]);
    }
    gidmask[idx] = any ? g : -1;
    unsigned long long bal = __ballot(any);
    if ((threadIdx.x & 31) == 0) {
        int c = __popcll(bal);
        if (c) atomicAdd(validCount, c);
    }
}

// ---------------- WMMA GEMM + streaming masked cross-entropy ----------------
__global__ __launch_bounds__(256)
void k_gemm_ce(const float* __restrict__ pred, const unsigned short* __restrict__ wbf,
               const float* __restrict__ bias, const int* __restrict__ gidmask,
               float* __restrict__ partials) {
    __shared__ unsigned short aT[ROWS_PER_BLOCK * AK];   // 66 KB normalized bf16 act
    __shared__ unsigned short wT[2][16 * AK];            // 16.5 KB dbl-buf weight tile
    __shared__ float wsum[WAVES];
    int tid = threadIdx.x, wave = tid >> 5, lane = tid & 31;
    int n16 = lane & 15, hi = lane >> 4;
    int rowsBase = blockIdx.x * ROWS_PER_BLOCK;

    // ---- stage A: each wave normalizes its own 16 rows to bf16 in LDS ----
    for (int r = 0; r < 16; ++r) {
        int lrow = wave * 16 + r;
        const float* src = pred + (size_t)(rowsBase + lrow) * EMBn + lane * 8;
        float4 f0 = *(const float4*)(src);
        float4 f1 = *(const float4*)(src + 4);
        float ss = f0.x*f0.x + f0.y*f0.y + f0.z*f0.z + f0.w*f0.w
                 + f1.x*f1.x + f1.y*f1.y + f1.z*f1.z + f1.w*f1.w;
        #pragma unroll
        for (int m = 16; m >= 1; m >>= 1) ss += __shfl_xor(ss, m, 32);
        float scale = EMB_SCALE / fmaxf(sqrtf(ss), 1e-12f);
        unsigned int p0 = (unsigned int)f2bf(f0.x*scale) | ((unsigned int)f2bf(f0.y*scale) << 16);
        unsigned int p1 = (unsigned int)f2bf(f0.z*scale) | ((unsigned int)f2bf(f0.w*scale) << 16);
        unsigned int p2 = (unsigned int)f2bf(f1.x*scale) | ((unsigned int)f2bf(f1.y*scale) << 16);
        unsigned int p3 = (unsigned int)f2bf(f1.z*scale) | ((unsigned int)f2bf(f1.w*scale) << 16);
        uint4 pk; pk.x = p0; pk.y = p1; pk.z = p2; pk.w = p3;
        *(uint4*)&aT[lrow * AK + lane * 8] = pk;
    }

    // per-lane rows: row = wave*16 + r + 8*hi  (C-matrix VGPR layout)
    int gidreg[8];
    float runmax[8], runsum[8], picked[8];
    #pragma unroll
    for (int r = 0; r < 8; ++r) {
        gidreg[r] = gidmask[rowsBase + wave * 16 + r + 8 * hi];
        runmax[r] = -INFINITY; runsum[r] = 0.f; picked[r] = -INFINITY;
    }

    // async DMA of one 16x256 bf16 weight tile straight into LDS (32B/thread)
    auto stage = [&](int j) {
        int n = tid >> 4, seg = tid & 15;
        unsigned short* dst = &wT[j & 1][n * AK + seg * 16];
        const unsigned short* src = wbf + (size_t)(j * 16 + n) * EMBn + seg * 16;
        unsigned d0 = (unsigned)(size_t)dst;      // low 32 bits = LDS offset
        asm volatile("global_load_async_to_lds_b128 %0, %1, off"
                     :: "v"(d0), "v"(src) : "memory");
        asm volatile("global_load_async_to_lds_b128 %0, %1, off"
                     :: "v"(d0 + 16u), "v"(src + 8) : "memory");
    };

    stage(0);
    wait_async0();
    __syncthreads();                              // also covers aT staging

    for (int j = 0; j < NT; ++j) {                // 63 class tiles
        if (j + 1 < NT) stage(j + 1);             // overlap DMA with WMMA
        const unsigned short* wTile = wT[j & 1];

        v8f c = {};
        #pragma unroll
        for (int kb = 0; kb < 8; ++kb) {
            int kblk = kb * 32;
            union { uint4 q[2]; v16bf v; } ua, ub;
            // A frag: elems 0..7 = K kblk+hi*8.. ; elems 8..15 = K kblk+16+hi*8..
            const unsigned short* ap = &aT[(wave * 16 + n16) * AK + kblk + hi * 8];
            ua.q[0] = *(const uint4*)(ap);
            ua.q[1] = *(const uint4*)(ap + 16);
            // B frag: lane col n16, elems 0..7 = K kblk+hi*8.. ; elems 8..15 = K kblk+16+hi*8..
            const unsigned short* bp = &wTile[n16 * AK + kblk + hi * 8];
            ub.q[0] = *(const uint4*)(bp);
            ub.q[1] = *(const uint4*)(bp + 16);
            c = __builtin_amdgcn_wmma_f32_16x16x32_bf16(
                    false, ua.v, false, ub.v, (short)0, c, false, false);
        }

        int ncls = j * 16 + n16;
        if (ncls < NIDn) {
            float bv = bias[ncls];
            #pragma unroll
            for (int r = 0; r < 8; ++r) {          // branchless streaming LSE
                float val = c[r] + bv;
                float nm  = fmaxf(runmax[r], val);
                runsum[r] = runsum[r] * __expf(runmax[r] - nm) + __expf(val - nm);
                runmax[r] = nm;
                picked[r] = (ncls == gidreg[r]) ? val : picked[r];
            }
        }

        wait_async0();                            // tile j+1 landed in LDS
        __syncthreads();
    }

    // ---- merge logsumexp state across the 16 lanes of each half-wave ----
    #pragma unroll
    for (int m = 1; m <= 8; m <<= 1) {
        #pragma unroll
        for (int r = 0; r < 8; ++r) {
            float om = __shfl_xor(runmax[r], m, 32);
            float os = __shfl_xor(runsum[r], m, 32);
            float op = __shfl_xor(picked[r], m, 32);
            float nm = fmaxf(runmax[r], om);
            runsum[r] = runsum[r] * __expf(runmax[r] - nm) + os * __expf(om - nm);
            runmax[r] = nm;
            picked[r] = fmaxf(picked[r], op);
        }
    }
    float local = 0.f;
    if (n16 == 0) {
        #pragma unroll
        for (int r = 0; r < 8; ++r)
            if (gidreg[r] >= 0)
                local += (runmax[r] + __logf(runsum[r])) - picked[r];
    }
    #pragma unroll
    for (int m = 16; m >= 1; m >>= 1) local += __shfl_xor(local, m, 32);
    if (lane == 0) wsum[wave] = local;
    __syncthreads();
    if (tid == 0) {
        float s = 0.f;
        #pragma unroll
        for (int w = 0; w < WAVES; ++w) s += wsum[w];
        partials[blockIdx.x] = s;                 // deterministic: no float atomics
    }
}

// ---------------- final deterministic reduction -----------------------------
__global__ __launch_bounds__(256)
void k_final(const float* __restrict__ partials, const int* __restrict__ validCount,
             float* __restrict__ out) {
    __shared__ float sh[256];
    int tid = threadIdx.x;
    float s = partials[tid] + partials[tid + 256] + partials[tid + 512] + partials[tid + 768];
    sh[tid] = s;
    __syncthreads();
    for (int st = 128; st > 0; st >>= 1) {
        if (tid < st) sh[tid] += sh[tid + st];
        __syncthreads();
    }
    if (tid == 0) out[0] = sh[0] / fmaxf((float)(*validCount), 1.0f);
}

extern "C" void kernel_launch(void* const* d_in, const int* in_sizes, int n_in,
                              void* d_out, int out_size, void* d_ws, size_t ws_size,
                              hipStream_t stream) {
    const float* conf   = (const float*)d_in[0];   // [B,P,C]
    const float* pred   = (const float*)d_in[1];   // [B,H,W,EMB]
    const int*   labels = (const int*)d_in[2];     // [B,P]
    const int*   gt     = (const int*)d_in[3];     // [B,H,W,A]
    const float* Wm     = (const float*)d_in[4];   // [NID,EMB]
    const float* bias   = (const float*)d_in[5];   // [NID]
    float* out = (float*)d_out;

    char* ws = (char*)d_ws;
    unsigned int*   keys     = (unsigned int*)ws;                 // 2 MB
    unsigned short* wbf      = (unsigned short*)(ws + 2097152);   // 504 KB
    int*            gidmask  = (int*)(ws + 2613248);              // 512 KB
    float*          partials = (float*)(ws + 3137536);            // 4 KB
    int*            cnt      = (int*)(ws + 3141632);
    int*            numPos     = cnt;        // 8
    int*            tieCnt     = cnt + 8;    // 8
    int*            validCount = cnt + 16;   // 1
    unsigned int*   Tarr       = (unsigned int*)(cnt + 17); // 8
    int*            Rarr       = cnt + 25;   // 8

    k_init  <<<1, 32, 0, stream>>>(cnt);
    k_bgloss<<<(Bn * Pn) / 256, 256, 0, stream>>>(conf, labels, keys, numPos);
    k_wconv <<<NPAD, 256, 0, stream>>>(Wm, wbf);
    k_radix <<<Bn, 1024, 0, stream>>>(keys, numPos, Tarr, Rarr);
    k_mask  <<<LOCn / 256, 256, 0, stream>>>(keys, labels, gt, Tarr, Rarr,
                                             tieCnt, gidmask, validCount);
    k_gemm_ce<<<LOCn / ROWS_PER_BLOCK, 256, 0, stream>>>(pred, wbf, bias,
                                                         gidmask, partials);
    k_final <<<1, 256, 0, stream>>>(partials, validCount, out);
}
